// CausalSelfAttention_79869211836707
// MI455X (gfx1250) — matrix operand order repaired and checked by reading, not compiled, not run on hardware
//
#include <hip/hip_runtime.h>
#include <hip/hip_bf16.h>

typedef __bf16 bf16_t;
typedef __attribute__((ext_vector_type(16))) __bf16 v16bf;
typedef __attribute__((ext_vector_type(8)))  __bf16 v8bf;
typedef __attribute__((ext_vector_type(4)))  __bf16 v4bf;
typedef __attribute__((ext_vector_type(8)))  float  v8f;

#define B_ 4
#define T_ 2048
#define D_ 1024
#define LDT 40                    // LDS row stride in bf16 elems (64B data + 16B pad)
#define TILE_ELEMS (128 * LDT)    // one 128x32 tile (padded) in bf16 elems
#define TILE_BYTES (TILE_ELEMS * 2)   // 10240 B

#if defined(__has_builtin)
#if __has_builtin(__builtin_amdgcn_tensor_load_to_lds) && __has_builtin(__builtin_amdgcn_s_wait_tensorcnt)
#define HAVE_TDM 1
#endif
#endif
#ifndef HAVE_TDM
#define HAVE_TDM 0
#endif

// ---- WMMA fragment helpers (ISA 7.12.2 layouts, wave32) ----

// A fragment: 16x32 bf16. Lanes 0-15: row=lane, K chunks [0..7],[16..23];
// lanes 16-31: row=lane-16, K chunks [8..15],[24..31].
__device__ __forceinline__ v16bf load_frag_a(const bf16_t* lds, int row0) {
    int lane = threadIdx.x & 31;
    int r    = lane & 15;
    int half = lane >> 4;
    const bf16_t* p = lds + (row0 + r) * LDT + half * 8;
    union { v16bf v; v8bf h[2]; } f;
    f.h[0] = *(const v8bf*)(p);
    f.h[1] = *(const v8bf*)(p + 16);
    return f.v;
}

// B fragment: 32x16 bf16 (16 N-rows x 32 contiguous K in LDS).
// Lanes 0-15: N=lane, K chunks [0..7],[16..23]; lanes 16-31: N=lane-16, K chunks [8..15],[24..31].
__device__ __forceinline__ v16bf load_frag_b(const bf16_t* lds, int n0) {
    int lane  = threadIdx.x & 31;
    int n     = lane & 15;
    int khalf = (lane >> 4) * 8;
    const bf16_t* p = lds + (n0 + n) * LDT + khalf;
    union { v16bf v; v8bf h[2]; } f;
    f.h[0] = *(const v8bf*)(p);
    f.h[1] = *(const v8bf*)(p + 16);
    return f.v;
}

__device__ __forceinline__ v8f wmma_bf16(v16bf a, v16bf b, v8f c) {
    return __builtin_amdgcn_wmma_f32_16x16x32_bf16(
        false, a, false, b, (short)0, c, false, false);
}

#if HAVE_TDM
typedef __attribute__((ext_vector_type(4))) unsigned int u32x4;
typedef __attribute__((ext_vector_type(8))) int i32x8;
typedef __attribute__((ext_vector_type(4))) int i32x4;

// TDM load of one 128-row x 32-elem bf16 tile into LDS with 16B row padding
// (pad_interval: 16 DWORDs -> enc 3, pad_amount: 4 DWORDs -> enc 3).
// D# per ISA 8.3/8.4: group0 {count, lds_addr, global_addr, type=2},
// group1 {data_size=2B, pad cfg, tensor_dim0/1, tile 32x128, dim0 stride}.
// This toolchain exposes the 6-arg builtin: (g0, g1, g2, g3, i32x8, cpol).
__device__ __forceinline__ void tdm_load_tile_128x32(
    unsigned ldsByteOff, const bf16_t* g, int strideElems, int tensorRows, int tensorK)
{
    unsigned long long ga = (unsigned long long)(const void*)g;
    u32x4 g0;
    g0[0] = 1u;                                           // count=1, user desc
    g0[1] = ldsByteOff;                                   // lds_addr
    g0[2] = (unsigned)ga;                                 // global_addr[31:0]
    g0[3] = (unsigned)((ga >> 32) & 0x01FFFFFFull)        // global_addr[56:32]
          | 0x80000000u;                                  // type=2 ("image")
    i32x8 g1;
    g1[0] = (int)(0x00010000u                             // data_size = 2 bytes
          | (1u << 20)                                    // pad_enable
          | (3u << 22)                                    // pad_interval = 16 DWORDs (64B)
          | (3u << 25));                                  // pad_amount   = 4 DWORDs (16B)
    g1[1] = (int)(((unsigned)tensorK & 0xFFFFu) << 16);   // tensor_dim0[15:0]
    g1[2] = (int)(((unsigned)tensorK >> 16)               // tensor_dim0[31:16]
          | (((unsigned)tensorRows & 0xFFFFu) << 16));    // tensor_dim1[15:0]
    g1[3] = (int)(((unsigned)tensorRows >> 16)            // tensor_dim1[31:16]
          | (32u << 16));                                 // tile_dim0 = 32
    g1[4] = 128;                                          // tile_dim1 = 128, tile_dim2 = 0
    g1[5] = strideElems;                                  // tensor_dim0_stride[31:0]
    g1[6] = 0;
    g1[7] = 0;
    i32x4 gz4 = {0, 0, 0, 0};
    i32x8 gz8 = {0, 0, 0, 0, 0, 0, 0, 0};
    __builtin_amdgcn_tensor_load_to_lds(g0, g1, gz4, gz4, gz8, 0);
}
#endif

// ---------------------------------------------------------------------------
// Kernel 1: Y[m,n] = sum_k X[m,k] * W[n,k]; fp32 in, bf16 out.
// TRANSPOSE=1 stores Y into vT[b][n][t] (b=m/T, t=m%T) for the P*V GEMM.
// Manual LDS staging (needs fp32->bf16 conversion in flight; TDM moves bytes).
// ---------------------------------------------------------------------------
template <int TRANSPOSE>
__global__ __launch_bounds__(256) void qkv_gemm_kernel(
    const float* __restrict__ X, const float* __restrict__ W,
    bf16_t* __restrict__ Y)
{
    __shared__ bf16_t sA[TILE_ELEMS];
    __shared__ bf16_t sB[TILE_ELEMS];

    const int t       = threadIdx.x;
    const int blkRow0 = blockIdx.y * 128;
    const int blkCol0 = blockIdx.x * 128;
    const int wave    = t >> 5;
    const int waveM   = wave & 3;     // 4 waves along M (32 rows each)
    const int waveN   = wave >> 2;    // 2 waves along N (64 cols each)

    v8f acc[2][4] = {};

    for (int k0 = 0; k0 < D_; k0 += 32) {
        __syncthreads();
#pragma unroll
        for (int i = 0; i < 4; ++i) {
            int idx = t + i * 256;
            int row = idx >> 3;
            int ch  = idx & 7;
            float4 fx = *(const float4*)(X + (long long)(blkRow0 + row) * D_ + k0 + ch * 4);
            float4 fw = *(const float4*)(W + (long long)(blkCol0 + row) * D_ + k0 + ch * 4);
            v4bf bx = { (bf16_t)fx.x, (bf16_t)fx.y, (bf16_t)fx.z, (bf16_t)fx.w };
            v4bf bw = { (bf16_t)fw.x, (bf16_t)fw.y, (bf16_t)fw.z, (bf16_t)fw.w };
            *(v4bf*)&sA[row * LDT + ch * 4] = bx;
            *(v4bf*)&sB[row * LDT + ch * 4] = bw;
        }
        __syncthreads();

        v16bf a[2], b[4];
#pragma unroll
        for (int tm = 0; tm < 2; ++tm) a[tm] = load_frag_a(sA, waveM * 32 + tm * 16);
#pragma unroll
        for (int tn = 0; tn < 4; ++tn) b[tn] = load_frag_b(sB, waveN * 64 + tn * 16);
#pragma unroll
        for (int tm = 0; tm < 2; ++tm)
#pragma unroll
            for (int tn = 0; tn < 4; ++tn)
                acc[tm][tn] = wmma_bf16(a[tm], b[tn], acc[tm][tn]);
    }

    const int lane = t & 31;
    const int n    = lane & 15;
    const int half = lane >> 4;
#pragma unroll
    for (int tm = 0; tm < 2; ++tm) {
#pragma unroll
        for (int tn = 0; tn < 4; ++tn) {
            int row0 = blkRow0 + waveM * 32 + tm * 16 + half * 8;
            int col  = blkCol0 + waveN * 64 + tn * 16 + n;
#pragma unroll
            for (int i = 0; i < 8; ++i) {
                int r = row0 + i;
                bf16_t val = (bf16_t)acc[tm][tn][i];
                if (TRANSPOSE) {
                    int b_ = r >> 11;          // r / T_
                    int tt = r & (T_ - 1);     // r % T_
                    Y[((long long)b_ * D_ + col) * T_ + tt] = val;
                } else {
                    Y[(long long)r * D_ + col] = val;
                }
            }
        }
    }
}

// ---------------------------------------------------------------------------
// Kernel 2/4: batched bf16 GEMM, O[m,n] = scale * sum_k A[m,k] * B[n,k].
// TDM double-buffered pipeline: wave0 DMAs tile k+1 while all waves compute k.
// The TDM builtin writes LDS invisibly to the compiler, so we escape the LDS
// pointers through an inline asm with a memory clobber each iteration; this
// keeps the ds_load fragment reads alive and ordered after the tensorcnt wait.
// ---------------------------------------------------------------------------
__global__ __launch_bounds__(256) void bf16_gemm_kernel(
    const bf16_t* __restrict__ A, const bf16_t* __restrict__ B,
    float* __restrict__ O,
    long long strideA, long long strideB, long long strideO,
    int lda, int ldb, int ldo, int K, int rowsA, int rowsB, float scale)
{
    __shared__ bf16_t smem[4 * TILE_ELEMS];   // [buf0:A,B][buf1:A,B] = 40 KB

    const int t       = threadIdx.x;
    const int blkRow0 = blockIdx.y * 128;
    const int blkCol0 = blockIdx.x * 128;
    const int wave    = t >> 5;
    const int waveM   = wave & 3;
    const int waveN   = wave >> 2;

    const bf16_t* gA = A + strideA * blockIdx.z + (long long)blkRow0 * lda;
    const bf16_t* gB = B + strideB * blockIdx.z + (long long)blkCol0 * ldb;
    float*        Ob = O + strideO * blockIdx.z;

#if HAVE_TDM
    const bool issuer = (t < 32);   // wave 0 drives the TDM
    if (issuer) {
        tdm_load_tile_128x32(0u,                   gA, lda, rowsA, K);
        tdm_load_tile_128x32((unsigned)TILE_BYTES, gB, ldb, rowsB, K);
    }
#endif

    v8f acc[2][4] = {};

    int cur = 0;
    for (int k0 = 0; k0 < K; k0 += 32, cur ^= 1) {
        const unsigned curOff = (unsigned)cur * (2u * TILE_ELEMS);   // elems
        bf16_t* sAcur = smem + curOff;
        bf16_t* sBcur = smem + curOff + TILE_ELEMS;
#if HAVE_TDM
        if (issuer) __builtin_amdgcn_s_wait_tensorcnt(0);
        __syncthreads();                       // tile[cur] visible; tile[cur^1] reads done
        // Tell the compiler the TDM may have written through these pointers.
        asm volatile("" : "+v"(sAcur), "+v"(sBcur) :: "memory");
        int k1 = k0 + 32;
        if (issuer && k1 < K) {                // prefetch next tile while computing
            const unsigned nxtByte = (unsigned)(cur ^ 1) * (2u * TILE_BYTES);
            tdm_load_tile_128x32(nxtByte,              gA + k1, lda, rowsA, K);
            tdm_load_tile_128x32(nxtByte + TILE_BYTES, gB + k1, ldb, rowsB, K);
        }
#else
        __syncthreads();
#pragma unroll
        for (int i = 0; i < 2; ++i) {
            int idx = t + i * 256;
            int row = idx >> 2;
            int ch  = idx & 3;
            *(uint4*)&sAcur[row * LDT + ch * 8] =
                *(const uint4*)(gA + (long long)row * lda + k0 + ch * 8);
            *(uint4*)&sBcur[row * LDT + ch * 8] =
                *(const uint4*)(gB + (long long)row * ldb + k0 + ch * 8);
        }
        __syncthreads();
#endif

        v16bf a[2], b[4];
#pragma unroll
        for (int tm = 0; tm < 2; ++tm) a[tm] = load_frag_a(sAcur, waveM * 32 + tm * 16);
#pragma unroll
        for (int tn = 0; tn < 4; ++tn) b[tn] = load_frag_b(sBcur, waveN * 64 + tn * 16);
#pragma unroll
        for (int tm = 0; tm < 2; ++tm)
#pragma unroll
            for (int tn = 0; tn < 4; ++tn)
                acc[tm][tn] = wmma_bf16(a[tm], b[tn], acc[tm][tn]);
    }

    const int lane = t & 31;
    const int n    = lane & 15;
    const int half = lane >> 4;
#pragma unroll
    for (int tm = 0; tm < 2; ++tm) {
#pragma unroll
        for (int tn = 0; tn < 4; ++tn) {
            int row0 = blkRow0 + waveM * 32 + tm * 16 + half * 8;
            int col  = blkCol0 + waveN * 64 + tn * 16 + n;
#pragma unroll
            for (int i = 0; i < 8; ++i)
                Ob[(long long)(row0 + i) * ldo + col] = acc[tm][tn][i] * scale;
        }
    }
}

// ---------------------------------------------------------------------------
// Kernel 3: row softmax over S (length T_ per row), bf16 out to P.
// ---------------------------------------------------------------------------
__global__ __launch_bounds__(256) void softmax_rows_kernel(
    const float* __restrict__ S, bf16_t* __restrict__ P)
{
    __shared__ float red[256];
    const long long row = blockIdx.x;
    const float* s = S + row * T_;
    bf16_t*      p = P + row * T_;
    const int t = threadIdx.x;

    float vals[8];
    float m = -3.0e38f;
#pragma unroll
    for (int i = 0; i < 8; ++i) {
        vals[i] = s[t + i * 256];
        m = fmaxf(m, vals[i]);
    }
    red[t] = m;
    __syncthreads();
    for (int off = 128; off > 0; off >>= 1) {
        if (t < off) red[t] = fmaxf(red[t], red[t + off]);
        __syncthreads();
    }
    m = red[0];
    __syncthreads();

    float sum = 0.0f;
#pragma unroll
    for (int i = 0; i < 8; ++i) {
        vals[i] = __expf(vals[i] - m);
        sum += vals[i];
    }
    red[t] = sum;
    __syncthreads();
    for (int off = 128; off > 0; off >>= 1) {
        if (t < off) red[t] += red[t + off];
        __syncthreads();
    }
    const float inv = 1.0f / red[0];

#pragma unroll
    for (int i = 0; i < 8; ++i)
        p[t + i * 256] = (bf16_t)(vals[i] * inv);
}

// ---------------------------------------------------------------------------

extern "C" void kernel_launch(void* const* d_in, const int* in_sizes, int n_in,
                              void* d_out, int out_size, void* d_ws, size_t ws_size,
                              hipStream_t stream) {
    (void)in_sizes; (void)n_in; (void)out_size; (void)ws_size;

    const float* x  = (const float*)d_in[0];
    const float* Wq = (const float*)d_in[1];
    const float* Wk = (const float*)d_in[2];
    const float* Wv = (const float*)d_in[3];
    float* out = (float*)d_out;

    char* ws = (char*)d_ws;
    const long long qkvBytes = (long long)B_ * T_ * D_ * 2;   // 16 MB each
    const long long sBytes   = (long long)B_ * T_ * T_ * 4;   // 64 MB
    bf16_t* qB = (bf16_t*)(ws);
    bf16_t* kB = (bf16_t*)(ws + qkvBytes);
    bf16_t* vT = (bf16_t*)(ws + 2 * qkvBytes);                // [B][D][T]
    float*  S  = (float*)(ws + 3 * qkvBytes);
    bf16_t* P  = (bf16_t*)(ws + 3 * qkvBytes + sBytes);

    dim3 blk(256);

    // 1) QKV projections: M = B*T = 8192, N = D = 1024.
    dim3 g1(D_ / 128, (B_ * T_) / 128);
    qkv_gemm_kernel<0><<<g1, blk, 0, stream>>>(x, Wq, qB);
    qkv_gemm_kernel<0><<<g1, blk, 0, stream>>>(x, Wk, kB);
    qkv_gemm_kernel<1><<<g1, blk, 0, stream>>>(x, Wv, vT);    // store transposed

    // 2) Scores: S[b] = (1/sqrt(D)) * q[b] @ k[b]^T   (M=N=T, K=D)
    dim3 g2(T_ / 128, T_ / 128, B_);
    bf16_gemm_kernel<<<g2, blk, 0, stream>>>(
        qB, kB, S,
        (long long)T_ * D_, (long long)T_ * D_, (long long)T_ * T_,
        D_, D_, T_, D_, T_, T_, 0.03125f);

    // 3) Row softmax -> bf16 P
    softmax_rows_kernel<<<B_ * T_, blk, 0, stream>>>(S, P);

    // 4) Output: O[b] = P[b] @ v[b]   (M=T, N=D, K=T; B operand = vT rows)
    dim3 g3(D_ / 128, T_ / 128, B_);
    bf16_gemm_kernel<<<g3, blk, 0, stream>>>(
        P, vT, out,
        (long long)T_ * T_, (long long)D_ * T_, (long long)T_ * D_,
        T_, T_, D_, T_, T_, D_, 1.0f);
}
